// StaticQuantizedLinear_44444321579337
// MI455X (gfx1250) — matrix operand order repaired and checked by reading, not compiled, not run on hardware
//
#include <hip/hip_runtime.h>

typedef __attribute__((ext_vector_type(16))) _Float16 v16h;
typedef __attribute__((ext_vector_type(8)))  _Float16 v8h;
typedef __attribute__((ext_vector_type(4)))  _Float16 v4h;
typedef __attribute__((ext_vector_type(8)))  float    v8f;

#define IN_F  4096
#define OUT_F 16384
#define M_TOK 128   // B*S = 8*16

// ---------------------------------------------------------------------------
// Kernel 1: convert x (f32 -> f16, row-major) and compute per-row sums of x
// (needed for exact zero-point handling: (q-zp)*s . x = s*(q.x) - s*zp*sum(x))
// ---------------------------------------------------------------------------
__global__ __launch_bounds__(256) void xprep_kernel(const float* __restrict__ x,
                                                    _Float16* __restrict__ xh,
                                                    float* __restrict__ rowsum) {
  __shared__ float sred[256];
  const int m = blockIdx.x;           // one block per token row
  const int t = threadIdx.x;
  const float4* xr = (const float4*)(x + (size_t)m * IN_F);
  float s = 0.f;
#pragma unroll
  for (int i = 0; i < IN_F / 4; i += 256) {     // 4 iterations of float4
    float4 v = xr[i + t];
    s += v.x + v.y + v.z + v.w;
    v4h h;
    h[0] = (_Float16)v.x; h[1] = (_Float16)v.y;
    h[2] = (_Float16)v.z; h[3] = (_Float16)v.w;
    *(v4h*)(xh + (size_t)m * IN_F + 4 * (i + t)) = h;
  }
  sred[t] = s;
  __syncthreads();
#pragma unroll
  for (int off = 128; off > 0; off >>= 1) {
    if (t < off) sred[t] += sred[t + off];
    __syncthreads();
  }
  if (t == 0) rowsum[m] = sred[0];
}

// ---------------------------------------------------------------------------
// Kernel 2: weight-streaming WMMA GEMM.
// One wave per 16-wide N-tile; each wave computes all 8 M-tiles (M=128), so
// every weight element is read from HBM exactly once (B-fragment reuse x8).
// ---------------------------------------------------------------------------
__global__ __launch_bounds__(32) void qlinear_wmma_kernel(
    const _Float16* __restrict__ xh,      // [128, 4096] f16
    const int*      __restrict__ w,       // [16384, 4096] int32 (int8-valued)
    const float*    __restrict__ scale_p, // scalar
    const float*    __restrict__ zp_p,    // scalar
    const float*    __restrict__ bias,    // [16384]
    const float*    __restrict__ rowsum,  // [128]
    float*          __restrict__ out) {   // [128, 16384]
  const int lane  = threadIdx.x & 31;
  const int r     = lane & 15;   // N column within tile (B/C/D), M row (A)
  const int hs    = lane >> 4;   // half-wave select (K-halves / M-halves)
  const int ntile = blockIdx.x;
  const int n     = ntile * 16 + r;

  const float scale = *scale_p;
  const float zp    = *zp_p;

  v8f acc[8];
#pragma unroll
  for (int t = 0; t < 8; ++t) acc[t] = 0.0f;

  // B source: weight row n; this lane covers K-offsets [hs*16, hs*16+16)
  const int*      wrow = w  + (size_t)n * IN_F + hs * 16;
  // A source: x row r; this lane covers K-offsets [hs*16, hs*16+16)
  const _Float16* xa   = xh + (size_t)r * IN_F + hs * 16;

#pragma unroll 2
  for (int k = 0; k < IN_F; k += 32) {
    // prefetch next K-chunk of the weight stream (global_prefetch)
    __builtin_prefetch(wrow + k + 32, 0, 1);

    // ---- B fragment: 16 consecutive int32 -> f16 (exact for |q|<=127) ----
    int4 q[4];
    q[0] = *(const int4*)(wrow + k + 0);
    q[1] = *(const int4*)(wrow + k + 4);
    q[2] = *(const int4*)(wrow + k + 8);
    q[3] = *(const int4*)(wrow + k + 12);
    v16h bh;
#pragma unroll
    for (int i = 0; i < 4; ++i) {
      bh[4 * i + 0] = (_Float16)q[i].x;
      bh[4 * i + 1] = (_Float16)q[i].y;
      bh[4 * i + 2] = (_Float16)q[i].z;
      bh[4 * i + 3] = (_Float16)q[i].w;
    }

    // ---- 8 M-tiles against the shared B fragment ----
#pragma unroll
    for (int t = 0; t < 8; ++t) {
      const _Float16* ap = xa + (size_t)t * 16 * IN_F + k;
      v8h lo = *(const v8h*)(ap);        // K: hs*16 .. hs*16+7
      v8h hi = *(const v8h*)(ap + 8);    // K: hs*16+8 .. hs*16+15
      v16h a;
#pragma unroll
      for (int i = 0; i < 8; ++i) { a[i] = lo[i]; a[i + 8] = hi[i]; }
      acc[t] = __builtin_amdgcn_wmma_f32_16x16x32_f16(
          /*neg_a=*/false, a, /*neg_b=*/false, bh,
          /*c_mod=*/(short)0, acc[t], /*reuse_a=*/false, /*reuse_b=*/false);
    }
  }

  // ---- epilogue: out = scale*acc - scale*zp*rowsum[m] + bias[n] ----
  const float bn  = bias[n];
  const float szp = scale * zp;
#pragma unroll
  for (int t = 0; t < 8; ++t) {
#pragma unroll
    for (int j = 0; j < 8; ++j) {
      const int m = t * 16 + hs * 8 + j;       // C/D layout: M = vgpr + 8*hs
      out[(size_t)m * OUT_F + n] = scale * acc[t][j] - szp * rowsum[m] + bn;
    }
  }
}

// ---------------------------------------------------------------------------
extern "C" void kernel_launch(void* const* d_in, const int* in_sizes, int n_in,
                              void* d_out, int out_size, void* d_ws, size_t ws_size,
                              hipStream_t stream) {
  const float* x     = (const float*)d_in[0];   // [8,16,4096] f32
  const int*   wint  = (const int*)  d_in[1];   // [16384,4096] int32
  const float* scale = (const float*)d_in[2];   // scalar
  const float* zp    = (const float*)d_in[3];   // scalar
  const float* bias  = (const float*)d_in[4];   // [16384]
  float*       out   = (float*)d_out;           // [8,16,16384]

  _Float16* xh     = (_Float16*)d_ws;                                    // 1 MB
  float*    rowsum = (float*)((char*)d_ws +
                              (size_t)M_TOK * IN_F * sizeof(_Float16));  // 512 B

  xprep_kernel<<<M_TOK, 256, 0, stream>>>(x, xh, rowsum);
  qlinear_wmma_kernel<<<OUT_F / 16, 32, 0, stream>>>(xh, wint, scale, zp,
                                                     bias, rowsum, out);
}
